// LMU_7799660610192
// MI455X (gfx1250) — matrix operand order repaired and checked by reading, not compiled, not run on hardware
//
#include <hip/hip_runtime.h>

// ---------------------------------------------------------------------------
// LMU recurrence for MI455X (gfx1250, wave32, WMMA).
//
// Batch-partitioned: 4 independent workgroups x 16 batch rows (M=16 WMMA
// tile), zero cross-WG sync. Per step, fused [x|h|m] (K=1024) bf16 WMMA GEMM
// against pre-packed N-major bf16 weights (L2-resident); AT held LDS-resident
// (320KB WGP LDS); m state fp32 in LDS; u via shuffle-reduction waves running
// in parallel with the WMMA waves. All WMMA loops manually double-buffered so
// fragment loads for iteration k+1 are in flight while WMMA k executes.
// ---------------------------------------------------------------------------

typedef __attribute__((ext_vector_type(16))) __bf16 bf16x16;
typedef __attribute__((ext_vector_type(8)))  __bf16 bf16x8;
typedef __attribute__((ext_vector_type(8)))  float  f32x8;

#define LMU_ORDER    256
#define LMU_UNITS    512
#define LMU_INDIM    256
#define LMU_BATCH    64
#define LMU_SEQ      2048
#define LMU_KTOT     1024           // INDIM + UNITS + ORDER
#define LMU_ASTRIDE  1032           // padded A row stride in bf16 elems (2064 B)
#define LMU_ATSTRIDE 264            // padded LDS AT row stride (528 B, 4-bank shift)

// ---------------------------------------------------------------------------
// Fragment helpers using contiguous k order (k = 16h + s) for both A and B.
// ---------------------------------------------------------------------------
__device__ __forceinline__ bf16x16 lmu_pack2(bf16x8 lo, bf16x8 hi) {
  union { bf16x16 v; struct { bf16x8 lo, hi; } p; } u;
  u.p.lo = lo; u.p.hi = hi;
  return u.v;
}

// A operand (16x32 bf16) from row-major LDS staging, stride LMU_ASTRIDE.
// lane<16: row=lane,   elems 0..15 = K kb..kb+15
// lane>=16: row=lane-16, elems 0..15 = K kb+16..kb+31 (kbase includes asel*8).
__device__ __forceinline__ bf16x16 lmu_a_frag(const __bf16* As, int arow, int kbase) {
  const int ks = (kbase & ~31) + ((kbase & 8) << 1);
  const bf16x8* p0 = (const bf16x8*)(As + arow * LMU_ASTRIDE + ks);
  const bf16x8* p1 = (const bf16x8*)(As + arow * LMU_ASTRIDE + ks + 8);
  return lmu_pack2(*p0, *p1);
}

// B operand (32x16 bf16) from N-major (transposed) global weight array.
// lane<16: col n0+lane, K kb..kb+15 ; lane>=16: col n0+lane-16, K kb+16..kb+31.
__device__ __forceinline__ bf16x16 lmu_b_frag(const __bf16* Wt, int ldk, int n0,
                                              int kb, int lane) {
  const int c = n0 + (lane & 15);
  const int k = kb + ((lane >> 4) << 4);
  return *(const bf16x16*)(Wt + (size_t)c * ldk + k);
}

// B operand from LDS-resident N-major array with padded stride.
__device__ __forceinline__ bf16x16 lmu_b_frag_lds(const __bf16* Wt, int n0,
                                                  int kb, int lane) {
  const int c = n0 + (lane & 15);
  const int k = kb + ((lane >> 4) << 4);
  const bf16x8* p0 = (const bf16x8*)(Wt + c * LMU_ATSTRIDE + k);
  const bf16x8* p1 = (const bf16x8*)(Wt + c * LMU_ATSTRIDE + k + 8);
  return lmu_pack2(*p0, *p1);
}

__device__ __forceinline__ f32x8 lmu_wmma(bf16x16 a, bf16x16 b, f32x8 c) {
  return __builtin_amdgcn_wmma_f32_16x16x32_bf16(
      /*neg_a=*/false, a, /*neg_b=*/false, b,
      /*c_mod=*/(short)0, c, /*reuse_a=*/false, /*reuse_b=*/false);
}

// ---------------------------------------------------------------------------
// Kernel 1: pack weights to bf16, N-major (transposed), concatenated K.
//   Wall_t[n][k], k in [0,256)->Wi[k][n], [256,768)->Wh[k-256][n],
//                 [768,1024)->Wm[k-768][n]          (512 x 1024 bf16, 1 MB)
//   AT_t[n][k] = AT[k][n]                            (256 x 256 bf16, 128 KB)
// ---------------------------------------------------------------------------
__global__ void lmu_pack_weights(const float* __restrict__ Wi,
                                 const float* __restrict__ Wh,
                                 const float* __restrict__ Wm,
                                 const float* __restrict__ AT,
                                 __bf16* __restrict__ Wall_t,
                                 __bf16* __restrict__ AT_t) {
  const int idx = blockIdx.x * 256 + threadIdx.x;
  const int n1 = LMU_UNITS * LMU_KTOT;          // 524288
  if (idx < n1) {
    const int n = idx >> 10;
    const int k = idx & 1023;
    float v;
    if (k < 256)       v = Wi[k * LMU_UNITS + n];
    else if (k < 768)  v = Wh[(k - 256) * LMU_UNITS + n];
    else               v = Wm[(k - 768) * LMU_UNITS + n];
    Wall_t[(size_t)n * LMU_KTOT + k] = (__bf16)v;
  } else if (idx < n1 + LMU_ORDER * LMU_ORDER) {
    const int j = idx - n1;
    const int n = j >> 8;
    const int k = j & 255;
    AT_t[(size_t)n * LMU_ORDER + k] = (__bf16)AT[k * LMU_ORDER + n];
  }
}

// ---------------------------------------------------------------------------
// Kernel 2: the sequential recurrence. grid = 4 blocks (16 batch rows each),
// block = 1024 threads = 32 waves (8 per SIMD32 for latency hiding).
//   wave w (0..31): h tile, cols [16w, 16w+16)           -> 32 WMMA / step
//   wave w (0..15): m tile, cols [16w, 16w+16)           ->  8 WMMA / step
//   wave w (16..31): u[w-16] dot product + shuffle reduce
// LDS: A staging 33KB + fp32 m 16KB + LDS-resident AT 132KB  ~181 KB (<320KB)
// ---------------------------------------------------------------------------
__global__ __launch_bounds__(1024, 1)
void lmu_recurrence(const float* __restrict__ x,
                    const float* __restrict__ ie,   // 256
                    const float* __restrict__ he,   // 512
                    const float* __restrict__ me,   // 256
                    const float* __restrict__ BTv,  // 256
                    const __bf16* __restrict__ Wall_t,
                    const __bf16* __restrict__ AT_t,
                    float* __restrict__ out) {
  __shared__ __bf16 Asm[16 * LMU_ASTRIDE];
  __shared__ float  mF[16 * LMU_ORDER];
  __shared__ float  uS[16];
  __shared__ __bf16 ATs[LMU_ORDER * LMU_ATSTRIDE];   // 132 KB, LDS-resident AT

  const int tid  = threadIdx.x;
  const int lane = tid & 31;
  const int wave = tid >> 5;
  const int bg   = blockIdx.x;          // batch group: rows [16*bg, 16*bg+16)

  const int arow = lane & 15;           // A-operand row for this lane
  const int asel = lane >> 4;           // 0/1: K-half select & C row-half
  const int ccol = lane & 15;           // C/D column within tile

  const int n0h = wave * 16;            // this wave's h output columns
  const int n0m = (wave & 15) * 16;     // this wave's m output columns

  // ---- zero-init state (h=0, m=0) and park AT in LDS (16B vector copies) ----
  for (int i = tid; i < 16 * LMU_ASTRIDE; i += 1024) Asm[i] = (__bf16)0.0f;
  for (int i = tid; i < 16 * LMU_ORDER;  i += 1024) mF[i]  = 0.0f;
  for (int i = tid; i < LMU_ORDER * (LMU_ORDER / 8); i += 1024) {
    const int n = i >> 5, c8 = (i & 31) * 8;
    *(bf16x8*)&ATs[n * LMU_ATSTRIDE + c8] =
        *(const bf16x8*)&AT_t[(size_t)n * LMU_ORDER + c8];
  }
  __syncthreads();

  for (int t = 0; t < LMU_SEQ; ++t) {
    // ------- stage 1: stage x_t (fp32 -> bf16) into A[:, 0:256) -------
    for (int i = tid; i < 16 * LMU_INDIM; i += 1024) {
      const int r = i >> 8, k = i & 255;
      const float v = x[(((size_t)(bg * 16 + r)) * LMU_SEQ + t) * LMU_INDIM + k];
      Asm[r * LMU_ASTRIDE + k] = (__bf16)v;
    }
    // prefetch next timestep's x rows (lowers to global_prefetch_b8)
    if (t + 1 < LMU_SEQ && tid < 16) {
      __builtin_prefetch(
          &x[(((size_t)(bg * 16 + tid)) * LMU_SEQ + (t + 1)) * LMU_INDIM], 0, 0);
    }
    __syncthreads();

    // ------- stage 2: (m-acc | u) in parallel, then h partial K<768 -------
    f32x8 macc = {0.f, 0.f, 0.f, 0.f, 0.f, 0.f, 0.f, 0.f};
    if (wave < 16) {
      // m @ AT from LDS-resident AT; double-buffered fragment pipeline
      bf16x16 amC = lmu_a_frag(Asm, arow, 768 + 0 + asel * 8);
      bf16x16 bmC = lmu_b_frag_lds(ATs, n0m, 0, lane);
#pragma unroll
      for (int kb = 32; kb < LMU_ORDER; kb += 32) {
        const bf16x16 amN = lmu_a_frag(Asm, arow, 768 + kb + asel * 8);
        const bf16x16 bmN = lmu_b_frag_lds(ATs, n0m, kb, lane);
        macc = lmu_wmma(amC, bmC, macc);
        amC = amN; bmC = bmN;
      }
      macc = lmu_wmma(amC, bmC, macc);
    } else {
      // u[row] = sum_k A_bf16[row][k] * enc[k], enc = [ie | he | me]
      const int row = wave - 16;
      float s = 0.0f;
      const int kbase = lane * 32;
#pragma unroll 8
      for (int kk = 0; kk < 32; ++kk) {
        const int k = kbase + kk;
        const float av = (float)Asm[row * LMU_ASTRIDE + k];
        const float ev = (k < 256) ? ie[k] : ((k < 768) ? he[k - 256] : me[k - 768]);
        s += av * ev;
      }
      for (int off = 16; off > 0; off >>= 1) s += __shfl_xor(s, off, 32);
      if (lane == 0) uS[row] = s;
    }
    // h partial accumulation over x|h regions (K in [0,768); old h, new x),
    // double-buffered so global B loads overlap WMMA issue.
    f32x8 acc = {0.f, 0.f, 0.f, 0.f, 0.f, 0.f, 0.f, 0.f};
    {
      bf16x16 aC = lmu_a_frag(Asm, arow, 0 + asel * 8);
      bf16x16 bC = lmu_b_frag(Wall_t, LMU_KTOT, n0h, 0, lane);
#pragma unroll
      for (int kb = 32; kb < 768; kb += 32) {
        const bf16x16 aN = lmu_a_frag(Asm, arow, kb + asel * 8);
        const bf16x16 bN = lmu_b_frag(Wall_t, LMU_KTOT, n0h, kb, lane);
        acc = lmu_wmma(aC, bC, acc);
        aC = aN; bC = bN;
      }
      acc = lmu_wmma(aC, bC, acc);
    }
    __syncthreads();

    // ------- stage 3: m_new = m_old + m@AT + u*BT  (fp32 state + bf16 stage)
    if (wave < 16) {
#pragma unroll
      for (int r = 0; r < 8; ++r) {
        const int row = r + asel * 8;
        const int c   = n0m + ccol;
        const float v = mF[row * LMU_ORDER + c] + macc[r] + uS[row] * BTv[c];
        mF[row * LMU_ORDER + c] = v;
        Asm[row * LMU_ASTRIDE + 768 + c] = (__bf16)v;
      }
    }
    __syncthreads();

    // ------- stage 4: finish h over new m (K in [768,1024)), tanh, store
    {
      bf16x16 aC = lmu_a_frag(Asm, arow, 768 + asel * 8);
      bf16x16 bC = lmu_b_frag(Wall_t, LMU_KTOT, n0h, 768, lane);
#pragma unroll
      for (int kb = 800; kb < LMU_KTOT; kb += 32) {
        const bf16x16 aN = lmu_a_frag(Asm, arow, kb + asel * 8);
        const bf16x16 bN = lmu_b_frag(Wall_t, LMU_KTOT, n0h, kb, lane);
        acc = lmu_wmma(aC, bC, acc);
        aC = aN; bC = bN;
      }
      acc = lmu_wmma(aC, bC, acc);
    }
#pragma unroll
    for (int r = 0; r < 8; ++r) {
      const int row = r + asel * 8;
      const int c   = n0h + ccol;
      const float hv = tanhf(acc[r]);
      Asm[row * LMU_ASTRIDE + 256 + c] = (__bf16)hv;                    // next-step A
      out[(((size_t)(bg * 16 + row)) * LMU_SEQ + t) * LMU_UNITS + c] = hv;
    }
    __syncthreads();
  }
}

// ---------------------------------------------------------------------------
// Host launch. Inputs (setup_inputs order, all fp32):
//  0:x 1:input_encoders 2:hidden_encoders 3:memory_encoders
//  4:input_kernel 5:hidden_kernel 6:memory_kernel 7:AT 8:BT
// Output: fp32 (64, 2048, 512). Workspace: 1 MB Wall_t + 128 KB AT_t.
// ---------------------------------------------------------------------------
extern "C" void kernel_launch(void* const* d_in, const int* in_sizes, int n_in,
                              void* d_out, int out_size, void* d_ws, size_t ws_size,
                              hipStream_t stream) {
  const float* x  = (const float*)d_in[0];
  const float* ie = (const float*)d_in[1];
  const float* he = (const float*)d_in[2];
  const float* me = (const float*)d_in[3];
  const float* Wi = (const float*)d_in[4];
  const float* Wh = (const float*)d_in[5];
  const float* Wm = (const float*)d_in[6];
  const float* AT = (const float*)d_in[7];
  const float* BT = (const float*)d_in[8];

  __bf16* Wall_t = (__bf16*)d_ws;
  __bf16* AT_t   = (__bf16*)((char*)d_ws + (size_t)LMU_UNITS * LMU_KTOT * sizeof(__bf16));

  // 524288 + 65536 elements to pack -> 2304 blocks of 256
  lmu_pack_weights<<<2304, 256, 0, stream>>>(Wi, Wh, Wm, AT, Wall_t, AT_t);

  // 4 independent batch groups of 16 rows; 1024 threads = 32 waves each
  lmu_recurrence<<<4, 1024, 0, stream>>>(x, ie, he, me, BT, Wall_t, AT_t,
                                         (float*)d_out);
}
